// Top2Router_50457275793418
// MI455X (gfx1250) — matrix operand order repaired and checked by reading, not compiled, not run on hardware
//
#include <hip/hip_runtime.h>
#include <hip/hip_bf16.h>
#include <math.h>

// ---------------------------------------------------------------------------
// Problem constants (from reference): B=16384 tokens, D=1024, E=8, H=4096
// cap = int(1.25 * (B*2/E)) = 5120 (multiple of 128)
// ---------------------------------------------------------------------------
#define NB     16384
#define ND     1024
#define NE     8
#define NH     4096
#define CAP    5120
#define CAPPAD 5120
#define NCHUNK 64          // token chunks of 256 for deterministic compaction

typedef __attribute__((ext_vector_type(16))) _Float16 v16h;
typedef __attribute__((ext_vector_type(8)))  _Float16 v8h;
typedef __attribute__((ext_vector_type(8)))  float    v8f;
typedef __attribute__((ext_vector_type(4)))  float    f4;
typedef __attribute__((ext_vector_type(4)))  int      i4;

union VU { v16h v; i4 q[2]; };
union PK { _Float16 h[2]; unsigned u; };

// ---------------------------------------------------------------------------
// Zero-fill kernels (harness poisons d_out / d_ws with 0xAA)
// ---------------------------------------------------------------------------
__global__ void zero_f32_kernel(float* p, int n4) {
    int i = blockIdx.x * blockDim.x + threadIdx.x;
    int stride = gridDim.x * blockDim.x;
    f4 z = {0.f, 0.f, 0.f, 0.f};
    for (; i < n4; i += stride) ((f4*)p)[i] = z;
}

__global__ void zero_i32_kernel(int* p, int n) {
    int i = blockIdx.x * blockDim.x + threadIdx.x;
    int stride = gridDim.x * blockDim.x;
    for (; i < n; i += stride) p[i] = 0;
}

// ---------------------------------------------------------------------------
// Router: one wave32 per token. logits = x[b] . Wr[e], softmax, top-2.
// ---------------------------------------------------------------------------
__global__ __launch_bounds__(256)
void router_kernel(const float* __restrict__ x, const float* __restrict__ Wr,
                   int* __restrict__ e1, int* __restrict__ e2,
                   float* __restrict__ w1, float* __restrict__ w2) {
    int wave = threadIdx.x >> 5;
    int lane = threadIdx.x & 31;
    int b = blockIdx.x * 8 + wave;
    if (b >= NB) return;

    const float* xr = x + (size_t)b * ND;
    float xv[32];
#pragma unroll
    for (int i = 0; i < 32; ++i) xv[i] = xr[i * 32 + lane];

    float logit[NE];
#pragma unroll
    for (int e = 0; e < NE; ++e) {
        const float* wr = Wr + e * ND;
        float p = 0.f;
#pragma unroll
        for (int i = 0; i < 32; ++i) p += xv[i] * wr[i * 32 + lane];
#pragma unroll
        for (int s = 16; s > 0; s >>= 1) p += __shfl_xor(p, s, 32);
        logit[e] = p;
    }

    if (lane == 0) {
        float m = logit[0];
#pragma unroll
        for (int e = 1; e < NE; ++e) m = fmaxf(m, logit[e]);
        float pe[NE], s = 0.f;
#pragma unroll
        for (int e = 0; e < NE; ++e) { pe[e] = __expf(logit[e] - m); s += pe[e]; }
        float inv = 1.f / s;
#pragma unroll
        for (int e = 0; e < NE; ++e) pe[e] *= inv;

        int i1 = 0; float p1 = pe[0];
#pragma unroll
        for (int e = 1; e < NE; ++e) if (pe[e] > p1) { p1 = pe[e]; i1 = e; }
        int i2 = (i1 == 0) ? 1 : 0; float p2 = pe[i2];
#pragma unroll
        for (int e = 0; e < NE; ++e)
            if (e != i1 && pe[e] > p2) { p2 = pe[e]; i2 = e; }

        e1[b] = i1; e2[b] = i2; w1[b] = p1; w2[b] = p2;
    }
}

// ---------------------------------------------------------------------------
// Deterministic per-expert compaction (count -> scan -> ordered write)
// ---------------------------------------------------------------------------
__global__ __launch_bounds__(256)
void count_kernel(const int* __restrict__ e1, const int* __restrict__ e2,
                  int* __restrict__ chunkCnt) {
    __shared__ int c[NE];
    if (threadIdx.x < NE) c[threadIdx.x] = 0;
    __syncthreads();
    int b = blockIdx.x * 256 + threadIdx.x;
    atomicAdd(&c[e1[b]], 1);
    atomicAdd(&c[e2[b]], 1);
    __syncthreads();
    if (threadIdx.x < NE) chunkCnt[threadIdx.x * NCHUNK + blockIdx.x] = c[threadIdx.x];
}

__global__ void scan_kernel(const int* __restrict__ chunkCnt,
                            int* __restrict__ chunkOff, int* __restrict__ cnt) {
    int e = threadIdx.x;
    if (e >= NE) return;
    int run = 0;
    for (int c = 0; c < NCHUNK; ++c) {
        chunkOff[e * NCHUNK + c] = run;
        run += chunkCnt[e * NCHUNK + c];
    }
    cnt[e] = run < CAP ? run : CAP;
}

__global__ void compact_kernel(const int* __restrict__ e1, const int* __restrict__ e2,
                               const float* __restrict__ w1, const float* __restrict__ w2,
                               const int* __restrict__ chunkOff,
                               int* __restrict__ tokList, float* __restrict__ wList) {
    if (threadIdx.x != 0) return;
    int base = blockIdx.x * 256;
    int off[NE];
#pragma unroll
    for (int e = 0; e < NE; ++e) off[e] = chunkOff[e * NCHUNK + blockIdx.x];
    for (int t = 0; t < 256; ++t) {
        int b = base + t;
        int a = e1[b]; int d = off[a]++;
        if (d < CAP) { tokList[a * CAPPAD + d] = b; wList[a * CAPPAD + d] = w1[b]; }
        a = e2[b]; d = off[a]++;
        if (d < CAP) { tokList[a * CAPPAD + d] = b; wList[a * CAPPAD + d] = w2[b]; }
    }
}

// ---------------------------------------------------------------------------
// Expert GEMM with v_wmma_f32_16x16x32_f16.
// Block: 256 threads = 8 waves (4 M x 2 N), tile 128x128, K step 64.
// PHASE 1: A = gathered x rows (f32 -> f16 via VGPRs), epilogue GELU -> h (f16)
// PHASE 2: A = h rows staged with GLOBAL_LOAD_ASYNC_TO_LDS_B128 (ASYNCcnt),
//          epilogue out[tok] += w * (y + b2)
// ---------------------------------------------------------------------------
#define LDA 72   // LDS row stride in halves (16B-aligned, bank-spread)

__device__ __forceinline__ float gelu_exact(float v) {
    return 0.5f * v * (1.0f + erff(v * 0.70710678118654752f));
}

template <int PHASE>
__global__ __launch_bounds__(256)
void expert_gemm_kernel(const float* __restrict__ xIn,      // PHASE 1 A source
                        const _Float16* __restrict__ hIn,   // PHASE 2 A source
                        const float* __restrict__ W,        // [K][N] f32 (expert slice)
                        const float* __restrict__ bias,     // [N]
                        const int* __restrict__ tokList,    // [CAPPAD] (expert slice)
                        const float* __restrict__ wList,    // [CAPPAD]
                        const int* __restrict__ cntPtr,
                        _Float16* __restrict__ hOut,        // PHASE 1 out
                        float* __restrict__ out,            // PHASE 2 out (accumulate)
                        int K, int N) {
    __shared__ __align__(16) _Float16 Asub[128 * LDA];
    __shared__ __align__(16) _Float16 Bsub[128 * LDA];
    __shared__ int   sTok[128];
    __shared__ float sW[128];

    const int m0 = blockIdx.x * 128;
    const int n0 = blockIdx.y * 128;
    const int cnt = cntPtr[0];
    if (m0 >= cnt) return;          // fully-padded tile: rows masked by w=0 downstream

    const int tid  = threadIdx.x;
    const int wave = tid >> 5, lane = tid & 31;
    const int wm = wave >> 1;       // 0..3 -> 32 rows each
    const int wn = wave & 1;        // 0..1 -> 64 cols each
    const int hi = lane >> 4, lm = lane & 15;

    if (tid < 128) {
        sTok[tid] = tokList[m0 + tid];
        sW[tid]   = wList[m0 + tid];
    }
    __syncthreads();

    // ---- per-thread A staging bases, hoisted out of the K loop ----
    const int rowS = tid >> 1;      // 2 threads per row, 32 halves each
    const int halfS = tid & 1;
    _Float16* const dstA = Asub + rowS * LDA + halfS * 32;
    const unsigned ldsA = (unsigned)(uintptr_t)dstA;           // wave-relative LDS addr
    const float*    srcA1 = nullptr;
    const _Float16* srcA2 = nullptr;
    if (PHASE == 1) srcA1 = xIn + (size_t)sTok[rowS] * ND + halfS * 32;
    else            srcA2 = hIn + (size_t)(m0 + rowS) * K + halfS * 32;

    v8f acc[2][4] = {};

    for (int ko = 0; ko < K; ko += 64) {
        __syncthreads();           // previous compute done -> LDS reusable
        // ---- stage A tile: 128 rows x 64 halves ----
        if (PHASE == 1) {
            const float* s = srcA1 + ko;
#pragma unroll
            for (int i = 0; i < 4; ++i) {
                f4 a = *(const f4*)(s + i * 8);
                f4 b = *(const f4*)(s + i * 8 + 4);
                v8h hv;
                hv[0] = (_Float16)a[0]; hv[1] = (_Float16)a[1];
                hv[2] = (_Float16)a[2]; hv[3] = (_Float16)a[3];
                hv[4] = (_Float16)b[0]; hv[5] = (_Float16)b[1];
                hv[6] = (_Float16)b[2]; hv[7] = (_Float16)b[3];
                *(v8h*)(dstA + i * 8) = hv;
            }
        } else {
            // CDNA5 async DMA: 64B per thread, global -> LDS, tracked on ASYNCcnt.
            // INST_OFFSET is added to both the LDS and the global address.
            const _Float16* g = srcA2 + ko;
            asm volatile(
                "global_load_async_to_lds_b128 %0, %1, off\n\t"
                "global_load_async_to_lds_b128 %0, %1, off offset:16\n\t"
                "global_load_async_to_lds_b128 %0, %1, off offset:32\n\t"
                "global_load_async_to_lds_b128 %0, %1, off offset:48"
                :: "v"(ldsA), "v"(g) : "memory");
        }
        // ---- stage B tile: 64 K-rows x 128 N-cols, transposed -> Bsub[N][K] ----
        // Each thread handles a K-row pair so LDS writes are packed b32 stores.
#pragma unroll
        for (int i = 0; i < 4; ++i) {
            int linear = i * 256 + tid;            // 0..1023
            int kr = linear >> 5;                  // K-row pair 0..31
            int nQ = linear & 31;                  // float4 column group
            const float* wp = W + (size_t)(ko + 2 * kr) * N + n0 + nQ * 4;
            f4 a = *(const f4*)wp;
            f4 b = *(const f4*)(wp + N);
            _Float16* dst = Bsub + (nQ * 4) * LDA + 2 * kr;
#pragma unroll
            for (int j = 0; j < 4; ++j) {
                PK pk;
                pk.h[0] = (_Float16)a[j];
                pk.h[1] = (_Float16)b[j];
                *(unsigned*)(dst + j * LDA) = pk.u;
            }
        }
        if (ko + 64 < K && tid < 64)               // prefetch next B K-rows
            __builtin_prefetch(W + (size_t)(ko + 64 + tid) * N + n0, 0, 1);
        if (PHASE == 2)
            asm volatile("s_wait_asynccnt 0x0" ::: "memory");
        __syncthreads();

        // ---- compute: 2 K-steps of 32 ----
#pragma unroll
        for (int kk = 0; kk < 2; ++kk) {
            const int k0 = kk * 32;
            v16h aF[2], bF[4];
#pragma unroll
            for (int mt = 0; mt < 2; ++mt) {
                // ISA 16-bit A 16x32 layout: lane lm row; hi selects K {0..7,16..23} vs {8..15,24..31}
                const _Float16* ab = Asub + (wm * 32 + mt * 16 + lm) * LDA + k0;
                VU u;
                u.q[0] = *(const i4*)(ab + hi * 8);
                u.q[1] = *(const i4*)(ab + 16 + hi * 8);
                aF[mt] = u.v;
            }
#pragma unroll
            for (int nt = 0; nt < 4; ++nt) {
                // 16-bit B 32x16 layout: lane lm col; hi selects K {0..7,16..23} vs {8..15,24..31}
                const _Float16* bb = Bsub + (wn * 64 + nt * 16 + lm) * LDA + k0 + hi * 8;
                VU u;
                u.q[0] = *(const i4*)(bb);
                u.q[1] = *(const i4*)(bb + 16);
                bF[nt] = u.v;
            }
#pragma unroll
            for (int mt = 0; mt < 2; ++mt)
#pragma unroll
                for (int nt = 0; nt < 4; ++nt)
                    acc[mt][nt] = __builtin_amdgcn_wmma_f32_16x16x32_f16(
                        false, aF[mt], false, bF[nt],
                        (short)0, acc[mt][nt], false, false);
        }
    }

    // ---- epilogue ----
    // C/D layout: VGPR r, lanes 0-15 -> (M=r, N=lm); lanes 16-31 -> (M=8+r, N=lm)
#pragma unroll
    for (int mt = 0; mt < 2; ++mt) {
#pragma unroll
        for (int nt = 0; nt < 4; ++nt) {
            const int col = n0 + wn * 64 + nt * 16 + lm;
            const float bv = bias[col];
#pragma unroll
            for (int r = 0; r < 8; ++r) {
                const int row = wm * 32 + mt * 16 + hi * 8 + r;
                float v = acc[mt][nt][r] + bv;
                if (PHASE == 1) {
                    hOut[(size_t)(m0 + row) * N + col] = (_Float16)gelu_exact(v);
                } else {
                    const float wgt = sW[row];
                    if (wgt > 0.f) {
                        float* p = out + (size_t)sTok[row] * N + col;
                        *p = *p + wgt * v;   // rows unique per expert; experts serialized
                    }
                }
            }
        }
    }
}

// ---------------------------------------------------------------------------
// Host launcher
// ---------------------------------------------------------------------------
extern "C" void kernel_launch(void* const* d_in, const int* in_sizes, int n_in,
                              void* d_out, int out_size, void* d_ws, size_t ws_size,
                              hipStream_t stream) {
    const float* x  = (const float*)d_in[0];   // [B, D]
    const float* Wr = (const float*)d_in[1];   // [E, D]
    const float* W1 = (const float*)d_in[2];   // [E, D, H]
    const float* b1 = (const float*)d_in[3];   // [E, H]
    const float* W2 = (const float*)d_in[4];   // [E, H, D]
    const float* b2 = (const float*)d_in[5];   // [E, D]
    float* out = (float*)d_out;                // [B, D]

    // Workspace layout
    int* e1 = (int*)d_ws;
    int* e2 = e1 + NB;
    float* w1v = (float*)(e2 + NB);
    float* w2v = w1v + NB;
    int* meta      = (int*)(w2v + NB);
    int* chunkCnt  = meta;                       // E*64
    int* chunkOff  = meta + NE * NCHUNK;         // E*64
    int* cnt       = meta + 2 * NE * NCHUNK;     // E (padded to 64)
    int* tokList   = meta + 2 * NE * NCHUNK + 64;          // E*CAPPAD
    float* wList   = (float*)(tokList + NE * CAPPAD);      // E*CAPPAD
    _Float16* h    = (_Float16*)(wList + NE * CAPPAD);     // CAPPAD*H (one expert)
    const int metaInts = 2 * NE * NCHUNK + 64 + 2 * NE * CAPPAD;

    // 1. zero output + routing metadata
    zero_f32_kernel<<<4096, 256, 0, stream>>>(out, (NB * ND) / 4);
    zero_i32_kernel<<<(metaInts + 255) / 256, 256, 0, stream>>>(meta, metaInts);

    // 2. router (softmax + top-2)
    router_kernel<<<NB / 8, 256, 0, stream>>>(x, Wr, e1, e2, w1v, w2v);

    // 3. deterministic compaction with capacity clamp
    count_kernel<<<NCHUNK, 256, 0, stream>>>(e1, e2, chunkCnt);
    scan_kernel<<<1, 32, 0, stream>>>(chunkCnt, chunkOff, cnt);
    compact_kernel<<<NCHUNK, 32, 0, stream>>>(e1, e2, w1v, w2v, chunkOff, tokList, wList);

    // 4. expert MLPs (sequential experts; h reused, stays resident in 192MB L2)
    for (int e = 0; e < NE; ++e) {
        const float* W1e = W1 + (size_t)e * ND * NH;
        const float* b1e = b1 + (size_t)e * NH;
        const float* W2e = W2 + (size_t)e * NH * ND;
        const float* b2e = b2 + (size_t)e * ND;
        const int* tokE = tokList + e * CAPPAD;
        const float* wE = wList + e * CAPPAD;

        dim3 g1(CAPPAD / 128, NH / 128);   // 40 x 32
        expert_gemm_kernel<1><<<g1, 256, 0, stream>>>(
            x, (const _Float16*)nullptr, W1e, b1e, tokE, wE, cnt + e,
            h, (float*)nullptr, ND, NH);

        dim3 g2(CAPPAD / 128, ND / 128);   // 40 x 8
        expert_gemm_kernel<2><<<g2, 256, 0, stream>>>(
            (const float*)nullptr, (const _Float16*)h, W2e, b2e, tokE, wE, cnt + e,
            (_Float16*)nullptr, out, NH, ND);
    }
}
